// Model_39676907881286
// MI455X (gfx1250) — compile-verified, matrix operand order checked by reading, not run on hardware
//
#include <hip/hip_runtime.h>
#include <hip/hip_bf16.h>

typedef __attribute__((ext_vector_type(16))) _Float16 v16h;
typedef __attribute__((ext_vector_type(4)))  _Float16 v4h;
typedef __attribute__((ext_vector_type(8)))  float    v8f;

constexpr int Bn = 4, Hn = 16, Sn = 2048, Dn = 128;
constexpr float INV_SCALE = 3.125f;   // 1 / 0.32

constexpr int KB    = 32;             // keys per block iteration
constexpr int QW    = 16;             // query rows per wave
constexpr int WAVES = 4;
constexpr int QB    = QW * WAVES;     // 64 query rows per workgroup

constexpr int KPAD = 8;               // half-element padding (16B) per K row
constexpr int VPAD = 8;               // half-element padding per V^T row

__global__ __launch_bounds__(WAVES * 32)
void fa2_wmma_kernel(const float* __restrict__ Q, const float* __restrict__ K,
                     const float* __restrict__ V, float* __restrict__ O)
{
    __shared__ _Float16 Ksh[KB][Dn + KPAD];     // K tile, f16 (8.5 KB)
    __shared__ _Float16 VTsh[Dn][KB + VPAD];    // V tile transposed, f16 (10 KB)
    __shared__ _Float16 Ps[WAVES][QW][KB];      // P staging (4 KB)

    const int bh     = blockIdx.y;
    const int qblock = blockIdx.x;
    const int tid    = threadIdx.x;
    const int wid    = tid >> 5;
    const int lane   = tid & 31;
    const int r      = lane & 15;   // row (A) / column (B) index within 16
    const int g      = lane >> 4;   // lane-half selector

    const size_t base = (size_t)bh * Sn * Dn;
    const float* Qb = Q + base;
    const float* Kb = K + base;
    const float* Vb = V + base;
    float*       Ob = O + base;

    const int q0 = qblock * QB + wid * QW;

    // ---- Q A-fragments (f16, pre-scaled by 1/0.32), 4 slabs of 32 d-values ----
    v16h qfrag[4];
    #pragma unroll
    for (int c = 0; c < 4; ++c) {
        const float* s0 = Qb + (size_t)(q0 + r) * Dn + c * 32 + g * 8;
        const float* s1 = s0 + 16;
        #pragma unroll
        for (int t = 0; t < 8; ++t) {
            qfrag[c][t]     = (_Float16)(s0[t] * INV_SCALE);
            qfrag[c][8 + t] = (_Float16)(s1[t] * INV_SCALE);
        }
    }

    // all-ones B fragment: row sums of P via one WMMA (P x 1 -> every col = row sum)
    v16h ones;
    #pragma unroll
    for (int t = 0; t < 16; ++t) ones[t] = (_Float16)1.0f;

    const v8f vzero = {};
    v8f acc[8];
    #pragma unroll
    for (int n = 0; n < 8; ++n) acc[n] = vzero;

    float m[8], l[8];
    #pragma unroll
    for (int v = 0; v < 8; ++v) { m[v] = -1e30f; l[v] = 0.0f; }

    constexpr int NBLK = Sn / KB;
    for (int kb = 0; kb < NBLK; ++kb) {
        const int k0 = kb * KB;
        __syncthreads();

        // ---- cooperative stage: convert ONCE to f16; K row-major, V transposed ----
        #pragma unroll
        for (int j = 0; j < 8; ++j) {
            const int idx = j * 128 + tid;     // 1024 float4 per tile
            const int row = idx >> 5;          // 32 float4 per row
            const int c4  = idx & 31;
            const float4 kq = *(const float4*)(Kb + (size_t)(k0 + row) * Dn + c4 * 4);
            v4h kh;
            kh[0] = (_Float16)kq.x; kh[1] = (_Float16)kq.y;
            kh[2] = (_Float16)kq.z; kh[3] = (_Float16)kq.w;
            *(v4h*)(&Ksh[row][c4 * 4]) = kh;   // 8B aligned -> ds_store_b64
            const float4 vv = *(const float4*)(Vb + (size_t)(k0 + row) * Dn + c4 * 4);
            VTsh[c4 * 4 + 0][row] = (_Float16)vv.x;
            VTsh[c4 * 4 + 1][row] = (_Float16)vv.y;
            VTsh[c4 * 4 + 2][row] = (_Float16)vv.z;
            VTsh[c4 * 4 + 3][row] = (_Float16)vv.w;
        }
        // prefetch next key/value tile while we compute on this one
        if (kb + 1 < NBLK) {
            const int prow = tid >> 2;              // 32 rows
            const int poff = (tid & 3) * 32;        // 4 x 128B per row
            __builtin_prefetch(Kb + (size_t)(k0 + KB + prow) * Dn + poff, 0, 1);
            __builtin_prefetch(Vb + (size_t)(k0 + KB + prow) * Dn + poff, 0, 1);
        }
        __syncthreads();

        // ---- preload ALL K B-fragments; fence so the loads cannot be re-sunk ----
        v16h kf[2][4];
        #pragma unroll
        for (int nt = 0; nt < 2; ++nt)
            #pragma unroll
            for (int c = 0; c < 4; ++c) {
                const _Float16* b0 = &Ksh[nt * 16 + r][c * 32 + g * 8];
                const _Float16* b1 = b0 + 16;
                #pragma unroll
                for (int t = 0; t < 8; ++t) {
                    kf[nt][c][t]     = b0[t];      // 8 halves = 16B -> ds_load_b128
                    kf[nt][c][8 + t] = b1[t];
                }
            }
        __builtin_amdgcn_sched_barrier(0);   // keep the 16 ds_loads batched ahead of WMMAs

        v8f sc[2];
        #pragma unroll
        for (int nt = 0; nt < 2; ++nt) {
            v8f s = vzero;
            #pragma unroll
            for (int c = 0; c < 4; ++c)
                s = __builtin_amdgcn_wmma_f32_16x16x32_f16(false, qfrag[c], false, kf[nt][c],
                                                           (short)0, s, false, false);
            sc[nt] = s;
        }

        // ---- online softmax: cross-lane max, exp; row sums done on the tensor pipe ----
        float alpha[8];
        #pragma unroll
        for (int v = 0; v < 8; ++v) {
            float t = fmaxf(sc[0][v], sc[1][v]);
            #pragma unroll
            for (int off = 1; off < 16; off <<= 1)
                t = fmaxf(t, __shfl_xor(t, off));
            const float mn = fmaxf(m[v], t);
            alpha[v] = __expf(m[v] - mn);
            sc[0][v] = __expf(sc[0][v] - mn);
            sc[1][v] = __expf(sc[1][v] - mn);
            m[v] = mn;
        }
        #pragma unroll
        for (int n = 0; n < 8; ++n)
            #pragma unroll
            for (int v = 0; v < 8; ++v)
                acc[n][v] *= alpha[v];

        // ---- P: C-layout -> LDS -> A-layout fragment ----
        #pragma unroll
        for (int v = 0; v < 8; ++v) {
            const int row = g * 8 + v;
            Ps[wid][row][r]      = (_Float16)sc[0][v];
            Ps[wid][row][16 + r] = (_Float16)sc[1][v];
        }
        asm volatile("s_wait_dscnt 0" ::: "memory");
        v16h pfrag;
        {
            const _Float16* p0 = &Ps[wid][r][g * 8];
            const _Float16* p1 = p0 + 16;
            #pragma unroll
            for (int t = 0; t < 8; ++t) {
                pfrag[t]     = p0[t];
                pfrag[8 + t] = p1[t];
            }
        }
        __builtin_amdgcn_sched_barrier(0);

        // ---- row sums on the tensor pipe: rs = P x ones ----
        const v8f rsv = __builtin_amdgcn_wmma_f32_16x16x32_f16(false, pfrag, false, ones,
                                                               (short)0, vzero, false, false);
        #pragma unroll
        for (int v = 0; v < 8; ++v)
            l[v] = l[v] * alpha[v] + rsv[v];

        // ---- preload ALL V B-fragments; fence; then O += P V (8 independent chains) ----
        v16h vf[8];
        #pragma unroll
        for (int n = 0; n < 8; ++n) {
            const _Float16* v0 = &VTsh[n * 16 + r][g * 8];
            const _Float16* v1 = v0 + 16;
            #pragma unroll
            for (int t = 0; t < 8; ++t) {
                vf[n][t]     = v0[t];
                vf[n][8 + t] = v1[t];
            }
        }
        __builtin_amdgcn_sched_barrier(0);   // keep the 16 ds_loads batched ahead of WMMAs

        #pragma unroll
        for (int n = 0; n < 8; ++n)
            acc[n] = __builtin_amdgcn_wmma_f32_16x16x32_f16(false, pfrag, false, vf[n],
                                                            (short)0, acc[n], false, false);
    }

    // ---- epilogue: normalize by row sums and store ----
    #pragma unroll
    for (int v = 0; v < 8; ++v) {
        const float inv = 1.0f / l[v];
        const int   row = q0 + g * 8 + v;
        float* dst = Ob + (size_t)row * Dn;
        #pragma unroll
        for (int n = 0; n < 8; ++n)
            dst[n * 16 + r] = acc[n][v] * inv;
    }
}

extern "C" void kernel_launch(void* const* d_in, const int* in_sizes, int n_in,
                              void* d_out, int out_size, void* d_ws, size_t ws_size,
                              hipStream_t stream) {
    (void)in_sizes; (void)n_in; (void)out_size; (void)d_ws; (void)ws_size;
    const float* Q = (const float*)d_in[0];
    const float* K = (const float*)d_in[1];
    const float* V = (const float*)d_in[2];
    float*       O = (float*)d_out;
    dim3 grid(Sn / QB, Bn * Hn);
    fa2_wmma_kernel<<<grid, WAVES * 32, 0, stream>>>(Q, K, V, O);
}
